// LSTM_15951508537668
// MI455X (gfx1250) — matrix operand order repaired and checked by reading, not compiled, not run on hardware
//
#include <hip/hip_runtime.h>
#include <stdint.h>

// LSTM forward for MI455X (gfx1250, wave32).
//  T=512, B=256, E=256, H=256.
//  Phase 0: pack Wx / Wh (4 fp32 gate matrices each) into bf16 WMMA-B tile layout.
//  Phase 1: x_proj = embeds @ Wx + b (bf16 WMMA, f32 accum), stored bf16 in the
//           WMMA C per-lane layout so both the store and the phase-2 reload are
//           single b128 ops.
//  Phase 2: 16 persistent workgroups (16 batch rows each), 512 serial steps.
//           h in LDS (bank-conflict-padded), c in WMMA accumulator registers.
// Workspace: 512KB Wx_packed + 512KB Wh_packed + 256MiB x_proj (bf16).

typedef __attribute__((ext_vector_type(16))) __bf16 v16bf;
typedef __attribute__((ext_vector_type(8)))  float  v8f;

#define T_DIM 512
#define B_DIM 256
#define E_DIM 256
#define H_DIM 256
#define G4H   1024
#define HPAD  264   // 256+8 bf16: row r -> bank offset 4r, conflict-free b128 A loads
#define FPAD  260   // fp32 out-staging row stride (dwords): rows 0..15 hit distinct banks

__device__ __forceinline__ float sigmoidf(float x) {
  return 1.0f / (1.0f + __expf(-x));
}

// ---------------------------------------------------------------------------
// Phase 0: pack concat([w0..w3], axis=1) (K=256 x N=1024 fp32) into the WMMA
// B-operand tile-contiguous bf16 layout:
//   packed[(((kt*64)+nt)*32 + lane)*16 + j] = W[kt*32 + (lane>>4)*16 + j][nt*16 + (lane&15)]
// so each lane's 16 B elements are 32 contiguous, 16B-aligned bytes.
// ---------------------------------------------------------------------------
__global__ void lstm_pack_w(const float* __restrict__ w0, const float* __restrict__ w1,
                            const float* __restrict__ w2, const float* __restrict__ w3,
                            __bf16* __restrict__ packed) {
  int g = blockIdx.x * blockDim.x + threadIdx.x;     // (kt,nt,lane) group id
  if (g >= 8 * 64 * 32) return;
  int lane = g & 31;
  int nt   = (g >> 5) & 63;
  int kt   = g >> 11;
  int n    = nt * 16 + (lane & 15);                  // 0..1023
  const float* w = (n < 256) ? w0 : (n < 512) ? w1 : (n < 768) ? w2 : w3;
  int nc    = n & 255;
  int kbase = kt * 32 + (lane >> 4) * 16;
  __bf16* dst = packed + (size_t)g * 16;
#pragma unroll
  for (int j = 0; j < 16; ++j)
    dst[j] = (__bf16)w[(kbase + j) * H_DIM + nc];    // native v_cvt_pk_bf16_f32, RNE
}

// ---------------------------------------------------------------------------
// Phase 1: x_proj = embeds[TB,256] @ Wx + b, stored bf16 in C-register layout:
//   xproj[((mt*64 + ntile)*32 + lane)*8 + r]   (one uint4 per lane per tile)
// Block = 8 waves (256 thr). Wave w: 16-row tile mt = bx*8+w, 4 n-tiles.
// ---------------------------------------------------------------------------
__global__ __launch_bounds__(256)
void lstm_xproj_gemm(const float* __restrict__ embeds,
                     const __bf16* __restrict__ wxp,
                     const float* __restrict__ pbg, const float* __restrict__ pbi,
                     const float* __restrict__ pbf, const float* __restrict__ pbo,
                     uint4* __restrict__ xproj) {
  const int lane = threadIdx.x & 31;
  const int wave = threadIdx.x >> 5;
  const int mt   = blockIdx.x * 8 + wave;            // 16-row tile, 0..8191
  const int n0   = blockIdx.y * 64;
  const int r15  = lane & 15;
  const int kgrp = lane >> 4;

  v8f zero = {0.f,0.f,0.f,0.f,0.f,0.f,0.f,0.f};
  v8f acc[4] = {zero, zero, zero, zero};

  const float* arow = embeds + (size_t)(mt * 16 + r15) * E_DIM;

  // t-invariant per-tile B base pointers (lane offset folded in).
  const uint4* bbase[4];
#pragma unroll
  for (int j = 0; j < 4; ++j)
    bbase[j] = (const uint4*)wxp + ((size_t)(blockIdx.y * 4 + j) * 32 + lane) * 2;

  for (int kt = 0; kt < 8; ++kt) {
    // ---- issue ALL loads for this k-step first (A raw fp32 + 4 B tiles) ----
    int ka = kt * 32 + kgrp * 16;
    float4 f0 = *(const float4*)(arow + ka);
    float4 f1 = *(const float4*)(arow + ka + 4);
    float4 f2 = *(const float4*)(arow + ka + 8);
    float4 f3 = *(const float4*)(arow + ka + 12);

    union { v16bf v; uint4 q[2]; } Bv[4];
#pragma unroll
    for (int j = 0; j < 4; ++j) {
      const uint4* bp = bbase[j] + (size_t)kt * 64 * 64;  // kt*64 tiles * 64 uint4
      Bv[j].q[0] = bp[0];
      Bv[j].q[1] = bp[1];
    }
    if (kt < 7)
      __builtin_prefetch((const void*)(arow + ka + 32), 0, 1);  // global_prefetch_b8

    // ---- convert A while B loads are in flight ----
    // A 16x32 bf16 tile, contiguous k (matches packed B): lane grp0 K 0..15 ; grp1 K 16..31
    union { v16bf v; __bf16 h[16]; } A;
    A.h[0]=(__bf16)f0.x; A.h[1]=(__bf16)f0.y; A.h[2]=(__bf16)f0.z; A.h[3]=(__bf16)f0.w;
    A.h[4]=(__bf16)f1.x; A.h[5]=(__bf16)f1.y; A.h[6]=(__bf16)f1.z; A.h[7]=(__bf16)f1.w;
    A.h[8]=(__bf16)f2.x; A.h[9]=(__bf16)f2.y; A.h[10]=(__bf16)f2.z; A.h[11]=(__bf16)f2.w;
    A.h[12]=(__bf16)f3.x; A.h[13]=(__bf16)f3.y; A.h[14]=(__bf16)f3.z; A.h[15]=(__bf16)f3.w;

    // ---- 4 WMMAs with staggered load waits ----
#pragma unroll
    for (int j = 0; j < 4; ++j)
      acc[j] = __builtin_amdgcn_wmma_f32_16x16x32_bf16(
          false, A.v, false, Bv[j].v, (short)0, acc[j], false, false);
  }

  // Epilogue: +bias, convert, one b128 store per tile (C layout preserved).
#pragma unroll
  for (int j = 0; j < 4; ++j) {
    int col  = n0 + j * 16 + r15;
    int gate = col >> 8;
    const float* bptr = (gate == 0) ? pbg : (gate == 1) ? pbi : (gate == 2) ? pbf : pbo;
    float bias = bptr[col & 255];
    union { uint4 q; __bf16 h[8]; } st;
#pragma unroll
    for (int r = 0; r < 8; ++r)
      st.h[r] = (__bf16)(acc[j][r] + bias);
    xproj[(size_t)(mt * 64 + (blockIdx.y * 4 + j)) * 32 + lane] = st.q;
  }
}

// ---------------------------------------------------------------------------
// Phase 2: recurrence. 16 blocks x 512 threads. Block b: batch rows b*16..+15.
// Wave w owns hidden slice w*16..+15 across all 4 gates (n-tiles
// {w, 16+w, 32+w, 48+w}) so the c/h update is purely register-local per wave.
// ---------------------------------------------------------------------------
__global__ __launch_bounds__(512)
void lstm_recurrence(const uint4* __restrict__ xproj,
                     const __bf16* __restrict__ whp,
                     float* __restrict__ out) {
  const int lane = threadIdx.x & 31;
  const int wave = threadIdx.x >> 5;       // 0..15 -> hidden slice
  const int b0   = blockIdx.x * 16;
  const int r15  = lane & 15;
  const int kgrp = lane >> 4;

  __shared__ __bf16 hbuf[16 * HPAD];       // h_t (bf16, padded rows)
  __shared__ float  hstg[16 * FPAD];       // h_t fp32 staging for coalesced out
  for (int i = threadIdx.x; i < 16 * HPAD; i += blockDim.x) hbuf[i] = (__bf16)0.0f;

  v8f zero = {0.f,0.f,0.f,0.f,0.f,0.f,0.f,0.f};
  v8f c = zero;
  const int ntile[4] = { wave, 16 + wave, 32 + wave, 48 + wave };

  // Time-invariant Wh tile base pointers (lane offset folded in).
  const uint4* bbase[4];
#pragma unroll
  for (int j = 0; j < 4; ++j)
    bbase[j] = (const uint4*)whp + ((size_t)ntile[j] * 32 + lane) * 2;

  const uint4* xrow = xproj + (size_t)blockIdx.x * 64 * 32 + lane;  // mt = t*16 + bx

  __syncthreads();                         // hbuf initialized

  for (int t = 0; t < T_DIM; ++t) {
    v8f acc[4] = {zero, zero, zero, zero};

    // x_proj tiles for this (t, batch block): issue early, consumed after k-loop.
    union { uint4 q; __bf16 h[8]; } xv[4];
#pragma unroll
    for (int j = 0; j < 4; ++j)
      xv[j].q = xrow[((size_t)t * 16 * 64 + (size_t)ntile[j]) * 32];

    for (int kt = 0; kt < 8; ++kt) {
      // A from LDS: ds_load_b128 x2, conflict-free (HPAD row stride).
      const uint4* av = (const uint4*)&hbuf[r15 * HPAD + kt * 32 + kgrp * 16];
      union { v16bf v; uint4 q[2]; } A;
      A.q[0] = av[0]; A.q[1] = av[1];

      // All 4 B tiles for this k-step up front -> staggered waits over 4 WMMAs.
      union { v16bf v; uint4 q[2]; } Bv[4];
#pragma unroll
      for (int j = 0; j < 4; ++j) {
        const uint4* bp = bbase[j] + (size_t)kt * 64 * 64;
        Bv[j].q[0] = bp[0];
        Bv[j].q[1] = bp[1];
      }
#pragma unroll
      for (int j = 0; j < 4; ++j)
        acc[j] = __builtin_amdgcn_wmma_f32_16x16x32_bf16(
            false, A.v, false, Bv[j].v, (short)0, acc[j], false, false);
    }

    float hn[8];
#pragma unroll
    for (int r = 0; r < 8; ++r) {
      float zg = acc[0][r] + (float)xv[0].h[r];
      float zi = acc[1][r] + (float)xv[1].h[r];
      float zf = acc[2][r] + (float)xv[2].h[r];
      float zo = acc[3][r] + (float)xv[3].h[r];
      float g  = tanhf(zg);
      float ig = sigmoidf(zi);
      float fg = sigmoidf(zf);
      float og = sigmoidf(zo);
      float cn = g * ig + c[r] * fg;
      c[r]  = cn;
      hn[r] = tanhf(cn) * og;
    }

    __syncthreads();                       // all reads of h_t complete
#pragma unroll
    for (int r = 0; r < 8; ++r) {
      int row = kgrp * 8 + r, col = wave * 16 + r15;
      hbuf[row * HPAD + col] = (__bf16)hn[r];
      hstg[row * FPAD + col] = hn[r];
    }
    __syncthreads();                       // h_{t+1} visible to all

    // Coalesced fp32 output: each thread stores 8 contiguous floats of the slab.
    {
      int e   = threadIdx.x * 8;           // 512*8 == 16*256
      int row = e >> 8, col = e & 255;
      const float* src = &hstg[row * FPAD + col];
      float4 a0 = *(const float4*)src;
      float4 a1 = *(const float4*)(src + 4);
      float* dst = out + ((size_t)t * B_DIM + b0 + row) * H_DIM + col;
      *(float4*)dst       = a0;
      *(float4*)(dst + 4) = a1;
    }
  }
}

// ---------------------------------------------------------------------------
extern "C" void kernel_launch(void* const* d_in, const int* in_sizes, int n_in,
                              void* d_out, int out_size, void* d_ws, size_t ws_size,
                              hipStream_t stream) {
  const float* embeds = (const float*)d_in[0];
  const float* W_gx = (const float*)d_in[1];
  const float* W_ix = (const float*)d_in[2];
  const float* W_fx = (const float*)d_in[3];
  const float* W_ox = (const float*)d_in[4];
  const float* W_gh = (const float*)d_in[5];
  const float* W_ih = (const float*)d_in[6];
  const float* W_fh = (const float*)d_in[7];
  const float* W_oh = (const float*)d_in[8];
  const float* b_g  = (const float*)d_in[9];
  const float* b_i  = (const float*)d_in[10];
  const float* b_f  = (const float*)d_in[11];
  const float* b_o  = (const float*)d_in[12];

  __bf16* ws    = (__bf16*)d_ws;
  __bf16* wxp   = ws;                      // 262144 bf16 (512 KB)
  __bf16* whp   = ws + 262144;             // 262144 bf16 (512 KB)
  uint4*  xproj = (uint4*)(ws + 524288);   // 512*256*1024 bf16 (256 MiB), C layout

  lstm_pack_w<<<64, 256, 0, stream>>>(W_gx, W_ix, W_fx, W_ox, wxp);
  lstm_pack_w<<<64, 256, 0, stream>>>(W_gh, W_ih, W_fh, W_oh, whp);

  // TB=131072 rows -> 8192 row-tiles -> grid.x = 1024 ; 4H=1024 -> grid.y = 16
  lstm_xproj_gemm<<<dim3(1024, 16), 256, 0, stream>>>(
      embeds, wxp, b_g, b_i, b_f, b_o, xproj);

  lstm_recurrence<<<16, 512, 0, stream>>>(xproj, whp, (float*)d_out);
}
